// MultiHeadAttention_48129403519279
// MI455X (gfx1250) — matrix operand order repaired and checked by reading, not compiled, not run on hardware
//
#include <hip/hip_runtime.h>

// ---------------------------------------------------------------------------
// MHA for MI455X (gfx1250): B=2, S=2048, D=1024, H=16, Dk=64
// Matrix math: v_wmma_f32_16x16x32_bf16 (wave32).
// Attention K/V staging: Tensor Data Mover (tensor_load_to_lds) double-buffered
// in LDS, tracked by TENSORcnt; fallback to cooperative copies if the builtin
// is unavailable.
// ---------------------------------------------------------------------------

typedef __bf16 bf16;
typedef __attribute__((ext_vector_type(16))) __bf16 v16bf;
typedef __attribute__((ext_vector_type(8)))  __bf16 v8bf;
typedef __attribute__((ext_vector_type(4)))  __bf16 v4bf;
typedef __attribute__((ext_vector_type(8)))  float  v8f;
typedef __attribute__((ext_vector_type(4)))  unsigned int u32x4;
typedef __attribute__((ext_vector_type(8)))  int i32x8;
typedef __attribute__((ext_vector_type(4)))  int i32x4;

#define WMMA_BF16(a, b, c) \
  __builtin_amdgcn_wmma_f32_16x16x32_bf16(false, (a), false, (b), (short)0, (c), false, false)

static constexpr int Bq = 2;
static constexpr int Sq = 2048;
static constexpr int Dm = 1024;
static constexpr int Hh = 16;
static constexpr int Dk = 64;
static constexpr int BS = Bq * Sq;  // 4096

// ---------------------------------------------------------------------------
// Tensor Data Mover support (probe-confirmed builtins, guarded).
// ---------------------------------------------------------------------------
#if defined(__HIP_DEVICE_COMPILE__) && \
    __has_builtin(__builtin_amdgcn_tensor_load_to_lds) && \
    __has_builtin(__builtin_amdgcn_s_wait_tensorcnt)
#define USE_TDM 1
#else
#define USE_TDM 0
#endif

#if USE_TDM
#if __has_include(<hip/amd_detail/amd_gfx1250_TDM.h>)
// amdgpu-toolchain / therock-10.0: 6-arg form
#define TDM_CALL(g0, g1) \
  __builtin_amdgcn_tensor_load_to_lds((g0), (g1), (i32x4)0, (i32x4)0, (i32x8)0, 0)
#else
// ROCm 7.2 clang-22: 5-arg form
#define TDM_CALL(g0, g1) \
  __builtin_amdgcn_tensor_load_to_lds((g0), (g1), (i32x4)0, (i32x4)0, 0)
#endif

// Issue a 2-D TDM load: tile_d0 contiguous elements per row, tile_d1 rows,
// row stride stride0 (elements), bf16 (2-byte) data. Tensor dims == tile dims
// so the transfer is always fully in-bounds.
__device__ inline void tdm_load_2d(unsigned lds_off, const void* gptr,
                                   unsigned tile_d0, unsigned tile_d1,
                                   unsigned long long stride0) {
  unsigned long long ga = (unsigned long long)(uintptr_t)gptr;
  u32x4 g0;
  g0[0] = 1u;                                           // count=1, user mode
  g0[1] = lds_off;                                      // lds_addr (bytes)
  g0[2] = (unsigned)(ga & 0xffffffffu);                 // global_addr[31:0]
  g0[3] = (unsigned)((ga >> 32) & 0x01ffffffu) | (2u << 30);  // [56:32] | type=2
  i32x8 g1;
  g1[0] = (int)(1u << 16);                              // data_size=1 (2B), mask=0
  g1[1] = (int)((tile_d0 & 0xffffu) << 16);             // tensor_dim0[15:0]
  g1[2] = (int)(((tile_d0 >> 16) & 0xffffu) |
                ((tile_d1 & 0xffffu) << 16));           // td0[31:16] | td1[15:0]
  g1[3] = (int)(((tile_d1 >> 16) & 0xffffu) |
                ((tile_d0 & 0xffffu) << 16));           // td1[31:16] | tile_dim0
  g1[4] = (int)(tile_d1 & 0xffffu);                     // tile_dim1, tile_dim2=0
  g1[5] = (int)(unsigned)(stride0 & 0xffffffffu);       // dim0_stride[31:0]
  g1[6] = (int)(unsigned)((stride0 >> 32) & 0xffffu);   // dim0_stride[47:32]
  g1[7] = 0;
  TDM_CALL(g0, g1);
}
#endif  // USE_TDM

// ---- fragment loaders -----------------------------------------------------
// A-matrix 16x32 bf16: kbase=(lane>>4)*16; elems 0..15 = row[kb..kb+15]
// (contiguous, matching ldB).
__device__ inline v16bf ldA(const bf16* p) {
  v8bf lo = *(const v8bf*)p;
  v8bf hi = *(const v8bf*)(p + 8);
  v16bf r;
#pragma unroll
  for (int i = 0; i < 8; ++i) { r[i] = lo[i]; r[i + 8] = hi[i]; }
  return r;
}
// B-matrix 32x16 bf16 (contiguous k order, matching ldA): lane half selects K half of 16;
// per-lane one contiguous 32B run.
__device__ inline v16bf ldB(const bf16* p) { return *(const v16bf*)p; }

// ---- fp32 -> bf16 conversion ----------------------------------------------
__global__ void cvt_kernel(const float* __restrict__ src, bf16* __restrict__ dst, int n4) {
  int i = blockIdx.x * blockDim.x + threadIdx.x;
  if (i < n4) {
    float4 f = ((const float4*)src)[i];
    v4bf o;
    o[0] = (bf16)f.x; o[1] = (bf16)f.y; o[2] = (bf16)f.z; o[3] = (bf16)f.w;
    ((v4bf*)dst)[i] = o;
  }
}

// ---- GEMM: OUT = (X[4096x1024] @ W^T[1024x1024] + bias) * scale ------------
// MODE 0: bf16 row-major.  MODE 1: bf16 transposed per batch Vt[b][1024][2048].
// MODE 2: fp32 row-major.
// Wave tile 64x64 (4x4 of 16x16); block 8 waves (2 M x 4 N) => 128x256.
template <int MODE>
__global__ __launch_bounds__(256) void gemm_xwt_kernel(
    const bf16* __restrict__ X, const bf16* __restrict__ W,
    const float* __restrict__ bias, void* __restrict__ outp, float scale) {
  const int lane = threadIdx.x & 31;
  const int wave = threadIdx.x >> 5;
  const int wm = wave & 1;
  const int wn = wave >> 1;
  const int m0 = blockIdx.x * 128 + wm * 64;
  const int n0 = blockIdx.y * 256 + wn * 64;
  const int lrow = lane & 15;
  const int hi16 = lane >> 4;
  const int kbA = hi16 * 16;
  const int kbB = hi16 * 16;

  v8f c[4][4] = {};

  const bf16* xrow[4];
  const bf16* wrow[4];
#pragma unroll
  for (int mi = 0; mi < 4; ++mi)
    xrow[mi] = X + (size_t)(m0 + mi * 16 + lrow) * Dm + kbA;
#pragma unroll
  for (int ni = 0; ni < 4; ++ni)
    wrow[ni] = W + (size_t)(n0 + ni * 16 + lrow) * Dm + kbB;

  for (int ks = 0; ks < Dm; ks += 32) {
    v16bf a[4];
#pragma unroll
    for (int mi = 0; mi < 4; ++mi) a[mi] = ldA(xrow[mi] + ks);
#pragma unroll
    for (int ni = 0; ni < 4; ++ni) {
      v16bf b = ldB(wrow[ni] + ks);
#pragma unroll
      for (int mi = 0; mi < 4; ++mi) c[mi][ni] = WMMA_BF16(a[mi], b, c[mi][ni]);
    }
  }

#pragma unroll
  for (int ni = 0; ni < 4; ++ni) {
    const int col = n0 + ni * 16 + lrow;
    const float bv = bias[col];
#pragma unroll
    for (int mi = 0; mi < 4; ++mi) {
#pragma unroll
      for (int r = 0; r < 8; ++r) {
        const int m = m0 + mi * 16 + r + 8 * hi16;
        const float v = (c[mi][ni][r] + bv) * scale;
        if (MODE == 0) {
          ((bf16*)outp)[(size_t)m * Dm + col] = (bf16)v;
        } else if (MODE == 1) {
          const int bb = m >> 11;
          const int ss = m & 2047;
          ((bf16*)outp)[(size_t)bb * Dm * Sq + (size_t)col * Sq + ss] = (bf16)v;
        } else {
          ((float*)outp)[(size_t)m * Dm + col] = v;
        }
      }
    }
  }
}

// ---- 16-lane row-group max reduction --------------------------------------
__device__ inline float rowmax16(float v) {
#pragma unroll
  for (int m = 1; m <= 8; m <<= 1) v = fmaxf(v, __shfl_xor(v, m, 32));
  return v;
}

// ---- causal flash attention -----------------------------------------------
// Q: [b][s][h*64+d] bf16 pre-scaled 1/8. K: same, unscaled. Vt: [b][d][s] bf16.
// Block = 4 waves handling one 64-row q block of one head; all waves share
// TDM-staged K/V windows (double-buffered). Grid = B*H*(S/64).
struct AttnSmem {
  alignas(16) bf16 K[2][32 * 64];   // [buf][kv_local][d]      8 KB
  alignas(16) bf16 V[2][64 * 32];   // [buf][d][kv_local]      8 KB
  alignas(16) bf16 P[4][16 * 32];   // per-wave P staging      4 KB
};

__global__ __launch_bounds__(128) void attn_kernel(
    const bf16* __restrict__ Q, const bf16* __restrict__ K,
    const bf16* __restrict__ Vt, bf16* __restrict__ ctx) {
  __shared__ AttnSmem sm;

  const int lane = threadIdx.x & 31;
  const int wave = threadIdx.x >> 5;
  const int tid  = threadIdx.x;
  const int nqb  = Sq / 64;  // 32
  const int bh   = blockIdx.x / nqb;
  const int qblk = blockIdx.x % nqb;
  const int b    = bh >> 4;
  const int h    = bh & 15;
  const int qb0   = qblk * 64;
  const int qbase = qb0 + wave * 16;
  const int kvend = qb0 + 64;  // block-uniform causal bound
  const int lrow = lane & 15;
  const int hi16 = lane >> 4;
  const int kbA = hi16 * 16;
  const int kbB = hi16 * 16;

  const bf16* Qp = Q + (size_t)b * Sq * Dm + (size_t)h * Dk;
  const bf16* Kp = K + (size_t)b * Sq * Dm + (size_t)h * Dk;
  const bf16* Vp = Vt + (size_t)b * Dm * Sq + (size_t)h * Dk * Sq;

  // Q A-fragments (resident for the whole kv loop)
  v16bf aq[2];
#pragma unroll
  for (int ks = 0; ks < 2; ++ks)
    aq[ks] = ldA(Qp + (size_t)(qbase + lrow) * Dm + ks * 32 + kbA);

  // all-ones B fragment: row sums via one extra WMMA instead of 32 bpermutes
  v16bf ones;
#pragma unroll
  for (int i = 0; i < 16; ++i) ones[i] = (bf16)1.0f;

  v8f o[4] = {};
  float mrow[8], lsum[8];
#pragma unroll
  for (int r = 0; r < 8; ++r) { mrow[r] = -1e30f; lsum[r] = 0.0f; }

  bf16* pl = &sm.P[wave][0];

  // ---- window staging: TDM (wave 0 issues DMA) or cooperative copy --------
  auto stage = [&](int buf, int kv0) {
#if USE_TDM
    if (wave == 0) {
      tdm_load_2d((unsigned)(uintptr_t)&sm.K[buf][0], Kp + (size_t)kv0 * Dm,
                  /*tile_d0=*/64, /*tile_d1=*/32, /*stride0=*/Dm);
      tdm_load_2d((unsigned)(uintptr_t)&sm.V[buf][0], Vp + kv0,
                  /*tile_d0=*/32, /*tile_d1=*/64, /*stride0=*/Sq);
    }
#else
    // 128 threads x 2 chunks x 16B cover each 4KB tile
    const int c0 = tid * 2;
#pragma unroll
    for (int i = 0; i < 2; ++i) {
      const int c = c0 + i;
      const int kr = c >> 3, ko = (c & 7) * 8;  // K: 32 rows x 8 chunks
      *(v8bf*)&sm.K[buf][kr * 64 + ko] =
          *(const v8bf*)(Kp + (size_t)(kv0 + kr) * Dm + ko);
      const int vr = c >> 2, vo = (c & 3) * 8;  // V: 64 rows x 4 chunks
      *(v8bf*)&sm.V[buf][vr * 32 + vo] =
          *(const v8bf*)(Vp + (size_t)vr * Sq + kv0 + vo);
    }
#endif
  };

  int cur = 0;
  stage(0, 0);  // prologue: window 0 in flight

  for (int kv0 = 0; kv0 < kvend; kv0 += 32) {
    const bool has_next = (kv0 + 32 < kvend);
    __syncthreads();  // previous compute done; other buffer free
    if (has_next) stage(cur ^ 1, kv0 + 32);
#if USE_TDM
    if (wave == 0) {  // current window's 2 TDM ops complete; next may overlap
      if (has_next) __builtin_amdgcn_s_wait_tensorcnt((short)2);
      else          __builtin_amdgcn_s_wait_tensorcnt((short)0);
    }
#endif
    __syncthreads();  // current buffer visible to all waves

    // ---- scores: two 16x16 tiles over kv columns kv0..kv0+31 (from LDS)
    v8f sc[2];
#pragma unroll
    for (int t = 0; t < 2; ++t) {
      const bf16* kp = &sm.K[cur][(t * 16 + lrow) * 64 + kbB];
      v16bf bk0 = ldB(kp);
      v16bf bk1 = ldB(kp + 32);
      v8f cc = {};
      cc = WMMA_BF16(aq[0], bk0, cc);
      cc = WMMA_BF16(aq[1], bk1, cc);
      sc[t] = cc;
    }

    // ---- causal mask (wave-uniform branch; covers fully-masked windows)
    if (kv0 + 31 > qbase) {
#pragma unroll
      for (int t = 0; t < 2; ++t) {
        const int kj = kv0 + t * 16 + lrow;
#pragma unroll
        for (int r = 0; r < 8; ++r) {
          const int qi = qbase + r + 8 * hi16;
          if (kj > qi) sc[t][r] = -1e30f;
        }
      }
    }

    // ---- online softmax: max via shuffles, exp, rescale O
    float alpha[8];
#pragma unroll
    for (int r = 0; r < 8; ++r) {
      float mx = rowmax16(fmaxf(sc[0][r], sc[1][r]));
      const float mn = fmaxf(mrow[r], mx);
      alpha[r] = __expf(mrow[r] - mn);
      mrow[r] = mn;
      sc[0][r] = __expf(sc[0][r] - mn);
      sc[1][r] = __expf(sc[1][r] - mn);
    }
#pragma unroll
    for (int dt = 0; dt < 4; ++dt)
#pragma unroll
      for (int r = 0; r < 8; ++r) o[dt][r] *= alpha[r];

    // ---- P: C-layout -> A-layout via per-wave LDS slice (in-order DS)
#pragma unroll
    for (int t = 0; t < 2; ++t)
#pragma unroll
      for (int r = 0; r < 8; ++r)
        pl[(r + 8 * hi16) * 32 + t * 16 + lrow] = (bf16)sc[t][r];

    v16bf pa = ldA(pl + lrow * 32 + kbA);

    // row sums = P @ ones (per-row results land in the C layout directly)
    v8f rs = {};
    rs = WMMA_BF16(pa, ones, rs);
#pragma unroll
    for (int r = 0; r < 8; ++r) lsum[r] = lsum[r] * alpha[r] + rs[r];

    // ---- O += P @ V (V tile from LDS, contiguous in kv)
#pragma unroll
    for (int dt = 0; dt < 4; ++dt) {
      v16bf bv = ldB(&sm.V[cur][(dt * 16 + lrow) * 32 + kbB]);
      o[dt] = WMMA_BF16(pa, bv, o[dt]);
    }

    cur ^= 1;
  }

  // ---- normalize and store ctx (bf16)
  float inv[8];
#pragma unroll
  for (int r = 0; r < 8; ++r) inv[r] = 1.0f / lsum[r];
#pragma unroll
  for (int dt = 0; dt < 4; ++dt) {
#pragma unroll
    for (int r = 0; r < 8; ++r) {
      const int s = qbase + r + 8 * hi16;
      ctx[((size_t)b * Sq + s) * Dm + h * Dk + dt * 16 + lrow] =
          (bf16)(o[dt][r] * inv[r]);
    }
  }
}

// ---------------------------------------------------------------------------
extern "C" void kernel_launch(void* const* d_in, const int* in_sizes, int n_in,
                              void* d_out, int out_size, void* d_ws, size_t ws_size,
                              hipStream_t stream) {
  (void)in_sizes; (void)n_in; (void)out_size; (void)ws_size;
  const float* q  = (const float*)d_in[0];
  const float* k  = (const float*)d_in[1];
  const float* v  = (const float*)d_in[2];
  // d_in[3] = mask: causal, handled analytically
  const float* Wq = (const float*)d_in[4];
  const float* bq = (const float*)d_in[5];
  const float* Wk = (const float*)d_in[6];
  const float* bk = (const float*)d_in[7];
  const float* Wv = (const float*)d_in[8];
  const float* bv = (const float*)d_in[9];
  const float* Wo = (const float*)d_in[10];
  const float* bo = (const float*)d_in[11];

  char* w = (char*)d_ws;
  const size_t MB = 1ull << 20;
  bf16* qb  = (bf16*)(w + 0 * MB);
  bf16* kb  = (bf16*)(w + 8 * MB);
  bf16* vb  = (bf16*)(w + 16 * MB);
  bf16* Qs  = (bf16*)(w + 24 * MB);
  bf16* Ks  = (bf16*)(w + 32 * MB);
  bf16* Vt  = (bf16*)(w + 40 * MB);
  bf16* ctx = (bf16*)(w + 48 * MB);
  bf16* Wqb = (bf16*)(w + 56 * MB);
  bf16* Wkb = (bf16*)(w + 58 * MB);
  bf16* Wvb = (bf16*)(w + 60 * MB);
  bf16* Wob = (bf16*)(w + 62 * MB);

  const int nAct4 = (BS * Dm) / 4;
  const int nWgt4 = (Dm * Dm) / 4;
  cvt_kernel<<<(nAct4 + 255) / 256, 256, 0, stream>>>(q, qb, nAct4);
  cvt_kernel<<<(nAct4 + 255) / 256, 256, 0, stream>>>(k, kb, nAct4);
  cvt_kernel<<<(nAct4 + 255) / 256, 256, 0, stream>>>(v, vb, nAct4);
  cvt_kernel<<<(nWgt4 + 255) / 256, 256, 0, stream>>>(Wq, Wqb, nWgt4);
  cvt_kernel<<<(nWgt4 + 255) / 256, 256, 0, stream>>>(Wk, Wkb, nWgt4);
  cvt_kernel<<<(nWgt4 + 255) / 256, 256, 0, stream>>>(Wv, Wvb, nWgt4);
  cvt_kernel<<<(nWgt4 + 255) / 256, 256, 0, stream>>>(Wo, Wob, nWgt4);

  dim3 ggrid(BS / 128, Dm / 256);  // (32, 4)
  gemm_xwt_kernel<0><<<ggrid, 256, 0, stream>>>(qb, Wqb, bq, (void*)Qs, 0.125f);
  gemm_xwt_kernel<0><<<ggrid, 256, 0, stream>>>(kb, Wkb, bk, (void*)Ks, 1.0f);
  gemm_xwt_kernel<1><<<ggrid, 256, 0, stream>>>(vb, Wvb, bv, (void*)Vt, 1.0f);

  attn_kernel<<<Bq * Hh * (Sq / 64), 128, 0, stream>>>(Qs, Ks, Vt, ctx);

  gemm_xwt_kernel<2><<<ggrid, 256, 0, stream>>>(ctx, Wob, bo, d_out, 1.0f);
}
